// CsmpPolicyModel_49306224558580
// MI455X (gfx1250) — matrix operand order repaired and checked by reading, not compiled, not run on hardware
//
#include <hip/hip_runtime.h>

typedef _Float16 v16h __attribute__((ext_vector_type(16)));
typedef _Float16 v8h  __attribute__((ext_vector_type(8)));
typedef float    v8f  __attribute__((ext_vector_type(8)));

#define WPB   4      // waves per block
#define NT_   100
#define NH_   40
#define H_    64

#define DT_     0.02f
#define A_C     0.1f
#define MU_     0.05f
#define SIGMA_  0.2f
#define PEN_    1.0f
#define EXPFAC_ 0.67032004603563930f   // exp(-LAMBD*NH*DT)
#define FINALD_ 0.81873075307798187f   // exp(-BETA*NT*DT)
#define DECAY_  0.99004983374916811f   // exp(-LAMBD*DT)
#define DSTEP_  0.99800199866933198f   // exp(-BETA*DT)
#define DD_     0.06f                  // DRIFT*(DRIFT+LAMBD)

__device__ __forceinline__ v8f wmma_f16(v16h a, v16h b, v8f c) {
  return __builtin_amdgcn_wmma_f32_16x16x32_f16(false, a, false, b, (short)0, c, false, false);
}
__device__ __forceinline__ v8f splat8(float v) {
  v8f r;
#pragma unroll
  for (int i = 0; i < 8; ++i) r[i] = v;
  return r;
}
__device__ __forceinline__ float sigm_(float v) { return 1.f / (1.f + __expf(-v)); }
__device__ __forceinline__ float tanh_(float v) { return 1.f - 2.f / (__expf(2.f * v) + 1.f); }
__device__ __forceinline__ unsigned short h2u(_Float16 h) { return __builtin_bit_cast(unsigned short, h); }
__device__ __forceinline__ _Float16 u2h(unsigned short u) { return __builtin_bit_cast(_Float16, u); }

// Value for packed B-fragment f, column col16, K = 16*hi + i within its 32-K chunk (i: K within 16-row group).
__device__ __forceinline__ float frag_val(int f, int hi, int col16, int i,
                                          const float* Wx, const float* Wh, const float* Wi) {
  if (f < 32) {                       // fused z/r gate weights, 4 K-chunks each
    int gate = f >> 4, j = (f >> 2) & 3, c = f & 3;
    int n = gate * 64 + j * 16 + col16;
    if (c < 2) { int kk = 32 * c + 16 * hi + i; return kk < NH_ ? Wx[kk * 192 + n] : 0.f; }
    int rr = 32 * (c - 2) + 16 * hi + i; return Wh[rr * 192 + n];
  } else if (f < 40) {                // g-gate x-part (Wx cols 128..191)
    int q = f - 32, j = q >> 1, c = q & 1;
    int n = 128 + j * 16 + col16;
    int kk = 32 * c + 16 * hi + i; return kk < NH_ ? Wx[kk * 192 + n] : 0.f;
  } else if (f < 48) {                // g-gate h-part (Wh cols 128..191)
    int q = f - 40, j = q >> 1, c = q & 1;
    int n = 128 + j * 16 + col16;
    int rr = 32 * c + 16 * hi + i; return Wh[rr * 192 + n];
  } else {                            // Winit (40x64, K padded to 64)
    int q = f - 48, j = q >> 1, c = q & 1;
    int n = j * 16 + col16;
    int kk = 32 * c + 16 * hi + i; return kk < NH_ ? Wi[kk * 64 + n] : 0.f;
  }
}

__global__ __launch_bounds__(WPB * 32)
void csmp_policy_kernel(const float* __restrict__ dw,
                        const float* __restrict__ x_init,
                        const float* __restrict__ wgt_in,
                        const float* __restrict__ Wx,
                        const float* __restrict__ Wh,
                        const float* __restrict__ bvec,
                        const float* __restrict__ Wo,
                        const float* __restrict__ bo,
                        const float* __restrict__ Wi,
                        const float* __restrict__ binit,
                        float* __restrict__ out) {
  __shared__ v16h      s_wf[56][32];        // packed B fragments (56 KB)
  __shared__ _Float16  s_hT[WPB][16][64];   // per-wave h transpose staging
  __shared__ float     s_wb[WPB][16][NH_];  // per-row de-scaled wgt circular buffer

  const int lane  = threadIdx.x & 31;
  const int wave  = threadIdx.x >> 5;
  const int col16 = lane & 15;
  const int hi    = lane >> 4;
  const int row   = (blockIdx.x * WPB + wave) * 16 + col16;

  // ---- pack weight fragments into LDS (once) ----
  for (int f = wave; f < 56; f += WPB) {
    v16h t;
#pragma unroll
    for (int i = 0; i < 16; ++i) t[i] = (_Float16)frag_val(f, i >> 3, col16, (i & 7) + 8 * hi, Wx, Wh, Wi);
    s_wf[f][lane] = t;
  }
  __syncthreads();

  // ---- per-row scalar state ----
  float x = x_init[row * NH_ + (NH_ - 1)];
  float S = 0.f;
#pragma unroll 8
  for (int j2 = 0; j2 < NH_; ++j2) {
    float u = wgt_in[row * NH_ + j2];
    if (!hi) s_wb[wave][col16][j2] = u;
    S += u;
  }
  float w0 = wgt_in[row * NH_ + 0];
  float wlast = wgt_in[row * NH_ + NH_ - 1];
  float gsc = 1.f, reward = 0.f, disc = 1.f;
  int head = 0;

  // ---- x_hist as WMMA A-fragments (K 0..63, zeros past 39) ----
  v16h xc0, xc1;
#pragma unroll
  for (int e = 0; e < 16; ++e) {
    int k0 = (e < 8) ? (hi * 8 + e) : ((hi ? 16 : 8) + e);
    int K0 = k0, K1 = 32 + k0;
    xc0[e] = (K0 < NH_) ? (_Float16)x_init[row * NH_ + K0] : (_Float16)0.f;
    xc1[e] = (K1 < NH_) ? (_Float16)x_init[row * NH_ + K1] : (_Float16)0.f;
  }

  // ---- per-column constants ----
  float bz[4], br[4], bg[4], wo[4], bi[4];
#pragma unroll
  for (int j = 0; j < 4; ++j) {
    bz[j] = bvec[j * 16 + col16];
    br[j] = bvec[64 + j * 16 + col16];
    bg[j] = bvec[128 + j * 16 + col16];
    wo[j] = Wo[j * 16 + col16];
    bi[j] = binit[j * 16 + col16];
  }
  float bo0 = bo[0];

  // ---- h0 = tanh(x_init @ Winit + binit) via WMMA ----
  v8f hc[4];
#pragma unroll
  for (int j = 0; j < 4; ++j) {
    v8f a = splat8(bi[j]);
    a = wmma_f16(xc0, s_wf[48 + j * 2 + 0][lane], a);
    a = wmma_f16(xc1, s_wf[48 + j * 2 + 1][lane], a);
#pragma unroll
    for (int v = 0; v < 8; ++v) hc[j][v] = tanh_(a[v]);
  }

  v16h hA2, hA3;
  auto stage_h = [&]() {
#pragma unroll
    for (int j = 0; j < 4; ++j)
#pragma unroll
      for (int v = 0; v < 8; ++v)
        s_hT[wave][v + 8 * hi][j * 16 + col16] = (_Float16)hc[j][v];
    asm volatile("s_wait_dscnt 0x0" ::: "memory");
    const _Float16* hrow = &s_hT[wave][col16][0];
    int gb = hi * 8;
    v8h p0 = *(const v8h*)(hrow + gb);
    v8h p1 = *(const v8h*)(hrow + gb + 16);
    v8h p2 = *(const v8h*)(hrow + gb + 32);
    v8h p3 = *(const v8h*)(hrow + gb + 48);
    hA2 = __builtin_shufflevector(p0, p1, 0,1,2,3,4,5,6,7,8,9,10,11,12,13,14,15);
    hA3 = __builtin_shufflevector(p2, p3, 0,1,2,3,4,5,6,7,8,9,10,11,12,13,14,15);
  };
  stage_h();

  for (int t = 0; t < NT_; ++t) {
    if (t > 0) {
      // shift x_hist fragments by one K, insert x at K=39
      unsigned u0 = (unsigned)h2u(xc0[0]) | ((unsigned)h2u(xc0[8]) << 16);
      unsigned u1 = (unsigned)h2u(xc1[0]) | ((unsigned)h2u(xc1[8]) << 16);
      unsigned r0 = (unsigned)__shfl_xor((int)u0, 16, 32);
      unsigned r1 = (unsigned)__shfl_xor((int)u1, 16, 32);
      _Float16 pf0 = u2h((unsigned short)(r0 & 0xffffu));
      _Float16 pf1 = u2h((unsigned short)(r0 >> 16));
      _Float16 pf2 = u2h((unsigned short)(r1 & 0xffffu));
      _Float16 pf3 = u2h((unsigned short)(r1 >> 16));
#pragma unroll
      for (int e = 0; e < 7; ++e) {
        xc0[e] = xc0[e + 1]; xc0[8 + e] = xc0[9 + e];
        xc1[e] = xc1[e + 1]; xc1[8 + e] = xc1[9 + e];
      }
      xc0[7]  = hi ? pf1 : pf0;
      xc0[15] = hi ? pf2 : pf1;
      xc1[7]  = hi ? pf3 : (_Float16)x;
      xc1[15] = hi ? (_Float16)0.f : pf3;
      // wgt transition: decay all (via scale), drop oldest, append x
      gsc *= DECAY_;
      S = (S - w0) * DECAY_ + x;
      if (!hi) s_wb[wave][col16][head] = x / gsc;
      head = head + 1; if (head == NH_) head = 0;
      w0 = s_wb[wave][col16][head] * gsc;
      wlast = x;
    }
    float zeta = (float)xc0[0];
    float y    = (S - 0.5f * (w0 + wlast)) * DT_;
    float xcm  = x + A_C * EXPFAC_ * y;

    // GRU: 48 WMMAs
#pragma unroll
    for (int j = 0; j < 4; ++j) {
      v8f az = splat8(bz[j]);
      az = wmma_f16(xc0, s_wf[j * 4 + 0][lane], az);
      az = wmma_f16(xc1, s_wf[j * 4 + 1][lane], az);
      az = wmma_f16(hA2, s_wf[j * 4 + 2][lane], az);
      az = wmma_f16(hA3, s_wf[j * 4 + 3][lane], az);
      v8f ar = splat8(br[j]);
      ar = wmma_f16(xc0, s_wf[16 + j * 4 + 0][lane], ar);
      ar = wmma_f16(xc1, s_wf[16 + j * 4 + 1][lane], ar);
      ar = wmma_f16(hA2, s_wf[16 + j * 4 + 2][lane], ar);
      ar = wmma_f16(hA3, s_wf[16 + j * 4 + 3][lane], ar);
      v8f agx = splat8(bg[j]);
      agx = wmma_f16(xc0, s_wf[32 + j * 2 + 0][lane], agx);
      agx = wmma_f16(xc1, s_wf[32 + j * 2 + 1][lane], agx);
      v8f agh = splat8(0.f);
      agh = wmma_f16(hA2, s_wf[40 + j * 2 + 0][lane], agh);
      agh = wmma_f16(hA3, s_wf[40 + j * 2 + 1][lane], agh);
#pragma unroll
      for (int v = 0; v < 8; ++v) {
        float z  = sigm_(az[v]);
        float r  = sigm_(ar[v]);
        float gg = tanh_(agx[v] + r * agh[v]);
        hc[j][v] = (1.f - z) * hc[j][v] + z * gg;
      }
    }

    // pi = h2 @ Wo + bo : per-lane dot + butterfly reduce + row select
    float pv[8];
#pragma unroll
    for (int v = 0; v < 8; ++v)
      pv[v] = hc[0][v] * wo[0] + hc[1][v] * wo[1] + hc[2][v] * wo[2] + hc[3][v] * wo[3];
#pragma unroll
    for (int m = 1; m < 16; m <<= 1)
#pragma unroll
      for (int v = 0; v < 8; ++v)
        pv[v] += __shfl_xor(pv[v], m, 32);
    int r3 = col16 & 7;
    float q = pv[0];
#pragma unroll
    for (int k = 1; k < 8; ++k) q = (r3 == k) ? pv[k] : q;
    float qq = __shfl_xor(q, 16, 32);
    float pi = ((((col16 < 8) ? 1 : 0) != hi) ? q : qq) + bo0;

    float util = 2.f * sqrtf(fmaxf(pi, 0.f) + 1e-12f);
    reward += (util * disc - fmaxf(-x, 0.f) * PEN_) * DT_;
    float dwt = dw[row * NT_ + t];
    float dx  = DD_ * y + MU_ * xcm + A_C * zeta - pi;
    x = x + dx * DT_ + SIGMA_ * xcm * dwt;
    disc *= DSTEP_;

    stage_h();  // h2 -> A-fragments for next step
  }

  // ---- final: one more shift, terminal utility ----
  {
    gsc *= DECAY_;
    S = (S - w0) * DECAY_ + x;
    if (!hi) s_wb[wave][col16][head] = x / gsc;
    head = head + 1; if (head == NH_) head = 0;
    w0 = s_wb[wave][col16][head] * gsc;
    wlast = x;
    float y   = (S - 0.5f * (w0 + wlast)) * DT_;
    float xcm = x + A_C * EXPFAC_ * y;
    reward += 2.f * sqrtf(fmaxf(xcm, 0.f) + 1e-12f) * FINALD_ - fmaxf(-x, 0.f) * PEN_;
    if (!hi) out[row] = -reward;
  }
}

extern "C" void kernel_launch(void* const* d_in, const int* in_sizes, int n_in,
                              void* d_out, int out_size, void* d_ws, size_t ws_size,
                              hipStream_t stream) {
  const float* dw   = (const float*)d_in[0];
  const float* x0   = (const float*)d_in[1];
  const float* wgt  = (const float*)d_in[2];
  const float* Wx   = (const float*)d_in[3];
  const float* Wh   = (const float*)d_in[4];
  const float* b    = (const float*)d_in[5];
  const float* Wo   = (const float*)d_in[6];
  const float* bo   = (const float*)d_in[7];
  const float* Wi   = (const float*)d_in[8];
  const float* bi   = (const float*)d_in[9];
  float* out = (float*)d_out;
  dim3 grid(8192 / (WPB * 16));
  dim3 block(WPB * 32);
  csmp_policy_kernel<<<grid, block, 0, stream>>>(dw, x0, wgt, Wx, Wh, b, Wo, bo, Wi, bi, out);
}
